// EdgeGAT_21303037788660
// MI455X (gfx1250) — matrix operand order repaired and checked by reading, not compiled, not run on hardware
//
#include <hip/hip_runtime.h>
#include <hip/hip_bf16.h>

#define NN 50000
#define EE 800000
#define HO 768          // fused output cols: f_ni | f_nj | h_src
#define LDS_STRIDE 264  // 256 + 8 pad (bf16 elems) -> conflict-free b128 LDS reads

typedef __attribute__((ext_vector_type(16))) __bf16 v16bf;
typedef __attribute__((ext_vector_type(8)))  float  v8f;

union BF16x16 { uint4 q[2]; v16bf v; unsigned short u[16]; };

static __device__ __forceinline__ unsigned short f2bf(float x) {
    unsigned int u = __float_as_uint(x);
    u += 0x7FFFu + ((u >> 16) & 1u);           // round-to-nearest-even
    return (unsigned short)(u >> 16);
}

static __device__ __forceinline__ void atomicMaxFloat(float* addr, float val) {
    if (val >= 0.f) atomicMax((int*)addr, __float_as_int(val));
    else            atomicMin((unsigned int*)addr, __float_as_uint(val));
}

// ---- prep: pack [256 x 768] fused weights into WMMA bf16 B-layout ------------
// wpack[((kt*48 + ct)*32 + lane)*16 + elem]; lane: l16=col-in-tile, hi=K half.
// B layout (16-bit, 32x16): lanes 0-15 col N=l16 K=(0..7,16..23), lanes 16-31 K=(8..15,24..31).
__global__ __launch_bounds__(256) void pack_w_kernel(
    const float* __restrict__ Wni, const float* __restrict__ Wnj,
    const float* __restrict__ Wnode, unsigned short* __restrict__ wpack) {
    int idx  = blockIdx.x * 256 + threadIdx.x;          // < 8*48*32*16 = 196608
    int elem = idx & 15;
    int lane = (idx >> 4) & 31;
    int rest = idx >> 9;
    int ct   = rest % 48;
    int kt   = rest / 48;
    int hi = lane >> 4, l16 = lane & 15;
    int k  = kt * 32 + (elem < 8 ? hi * 8 + elem : 16 + hi * 8 + (elem - 8)); // 0..255
    int ng = ct * 16 + l16;                             // 0..767
    const float* W = (ng < 256) ? Wni : (ng < 512 ? Wnj : Wnode);
    int n = ng & 255;
    wpack[idx] = f2bf(W[k * 256 + n]);
}

// wsum[j] = sum_i W_fij[i][j]  (efeats @ W_fij == reward * wsum)
__global__ void wsum_kernel(const float* __restrict__ Wfij, float* __restrict__ wsum) {
    int j = threadIdx.x;                                // 256 threads, 1 block
    float s = 0.f;
    #pragma unroll
    for (int i = 0; i < 32; ++i) s += Wfij[i * 256 + j];
    wsum[j] = s;
}

__global__ __launch_bounds__(256) void init_kernel(float* __restrict__ emax,
                                                   float* __restrict__ denom,
                                                   float* __restrict__ hout) {
    int t = blockIdx.x * 256 + threadIdx.x;             // < N*256
    hout[t] = 0.f;
    if (t < NN * 4) {
        denom[t] = 0.f;
        emax[t]  = __int_as_float(0xFF800000);          // -inf
    }
}

// ---- fused projection GEMM: Fbig[N,768] = bf16(nfeats) @ bf16(W_fused) -------
// grid = (3125 row-tiles, 3 col-groups), block = 256 (8 waves).
// Each wave owns TWO adjacent 16-col tiles -> two independent WMMA accumulator
// chains, so consecutive v_wmma ops have no RAW hazard (no v_nop padding).
__global__ __launch_bounds__(256) void gemm_kernel(
    const float* __restrict__ nfeats, const unsigned short* __restrict__ wpack,
    float* __restrict__ Fbig) {
    __shared__ __align__(16) unsigned short As[16 * LDS_STRIDE];
    const int m0   = blockIdx.x * 16;
    const int wave = threadIdx.x >> 5;
    const int lane = threadIdx.x & 31;
    // stage A tile (16 rows x 256 K) into LDS, fp32 -> bf16
    for (int i = threadIdx.x; i < 16 * 256; i += 256) {
        int rr = i >> 8, kk = i & 255;
        As[rr * LDS_STRIDE + kk] = f2bf(nfeats[(size_t)(m0 + rr) * 256 + kk]);
    }
    __syncthreads();
    const int ct0 = blockIdx.y * 16 + wave * 2;         // even tile id, 0..46
    const int l16 = lane & 15, hi = lane >> 4;
    v8f c0 = {};
    v8f c1 = {};
    #pragma unroll
    for (int kt = 0; kt < 8; ++kt) {
        // A fragment: lanes 0-15 K=(0..7,16..23), lanes 16-31 K=(8..15,24..31)
        const unsigned short* ap = &As[l16 * LDS_STRIDE + kt * 32 + hi * 8];
        BF16x16 a;
        a.q[0] = *(const uint4*)(ap);
        a.q[1] = *(const uint4*)(ap + 16);
        // B fragments for the two column tiles (32B contiguous per lane each)
        const unsigned short* bp0 = wpack + ((size_t)((kt * 48 + ct0) * 32 + lane) << 4);
        BF16x16 b0, b1;
        b0.q[0] = *(const uint4*)(bp0);
        b0.q[1] = *(const uint4*)(bp0 + 8);
        b1.q[0] = *(const uint4*)(bp0 + 512);           // next tile = +32*16 elems
        b1.q[1] = *(const uint4*)(bp0 + 520);
        c0 = __builtin_amdgcn_wmma_f32_16x16x32_bf16(false, a.v, false, b0.v,
                                                     (short)0, c0, false, false);
        c1 = __builtin_amdgcn_wmma_f32_16x16x32_bf16(false, a.v, false, b1.v,
                                                     (short)0, c1, false, false);
    }
    // C/D layout: VGPR r -> row M=r (lanes 0-15) / M=r+8 (lanes 16-31), col=l16
    float* out0 = Fbig + (size_t)(m0 + hi * 8) * HO + ct0 * 16 + l16;
    #pragma unroll
    for (int r = 0; r < 8; ++r) {
        out0[(size_t)r * HO]      = c0[r];
        out0[(size_t)r * HO + 16] = c1[r];
    }
}

// ---- edge attention logits + segment max (1 wave / edge) ---------------------
__global__ __launch_bounds__(256) void logits_kernel(
    const float* __restrict__ F, const float* __restrict__ reward,
    const int* __restrict__ src, const int* __restrict__ dst,
    const float* __restrict__ wsum, const float* __restrict__ bias,
    const float* __restrict__ attn, float* __restrict__ elog,
    float* __restrict__ emax) {
    int e = blockIdx.x * 8 + (threadIdx.x >> 5);
    if (e >= EE) return;
    int lane = threadIdx.x & 31;
    int s = src[e], d = dst[e];
    float r = reward[e];
    int j0 = lane * 8;
    const float4* pni = (const float4*)(F + (size_t)s * HO + j0);         // f_ni
    const float4* pnj = (const float4*)(F + (size_t)d * HO + 256 + j0);   // f_nj
    float4 a0 = pni[0], a1 = pni[1];
    float4 b0 = pnj[0], b1 = pnj[1];
    float va[8] = {a0.x, a0.y, a0.z, a0.w, a1.x, a1.y, a1.z, a1.w};
    float vb[8] = {b0.x, b0.y, b0.z, b0.w, b1.x, b1.y, b1.z, b1.w};
    float acc = 0.f;
    #pragma unroll
    for (int t = 0; t < 8; ++t) {
        int j = j0 + t;
        float v = va[t] + vb[t] + r * wsum[j] + bias[j];
        v = v > 0.f ? v : 0.2f * v;                      // leaky_relu
        acc += v * attn[j];                              // attn flat [H*O]
    }
    acc += __shfl_xor(acc, 1);
    acc += __shfl_xor(acc, 2);
    acc += __shfl_xor(acc, 4);                           // reduce 8 lanes / head
    if ((lane & 7) == 0) {
        int h = lane >> 3;
        elog[(size_t)e * 4 + h] = acc;
        atomicMaxFloat(&emax[(size_t)d * 4 + h], acc);
    }
}

// ---- exp(e - max) + segment sum (1 thread / (edge,head)) ---------------------
__global__ __launch_bounds__(256) void expsum_kernel(
    float* __restrict__ elog, const float* __restrict__ emax,
    const int* __restrict__ dst, float* __restrict__ denom) {
    int t = blockIdx.x * 256 + threadIdx.x;              // < E*4
    int e = t >> 2, h = t & 3;
    int d = dst[e];
    float x = __expf(elog[t] - emax[(size_t)d * 4 + h]);
    elog[t] = x;                                         // overwrite with ee
    atomicAdd(&denom[(size_t)d * 4 + h], x);
}

// ---- weighted scatter-sum of h_src (1 wave / edge) ---------------------------
__global__ __launch_bounds__(256) void aggregate_kernel(
    const float* __restrict__ F, const float* __restrict__ ee,
    const float* __restrict__ denom, const int* __restrict__ src,
    const int* __restrict__ dst, float* __restrict__ hout) {
    int e = blockIdx.x * 8 + (threadIdx.x >> 5);
    if (e >= EE) return;
    int lane = threadIdx.x & 31;
    int s = src[e], d = dst[e];
    int j0 = lane * 8, h = lane >> 3;
    float a = ee[(size_t)e * 4 + h] / denom[(size_t)d * 4 + h];
    const float4* ph = (const float4*)(F + (size_t)s * HO + 512 + j0);    // h_src
    float4 h0 = ph[0], h1 = ph[1];
    float* o = hout + (size_t)d * 256 + j0;
    atomicAdd(o + 0, a * h0.x); atomicAdd(o + 1, a * h0.y);
    atomicAdd(o + 2, a * h0.z); atomicAdd(o + 3, a * h0.w);
    atomicAdd(o + 4, a * h1.x); atomicAdd(o + 5, a * h1.y);
    atomicAdd(o + 6, a * h1.z); atomicAdd(o + 7, a * h1.w);
}

// ---- head mean + ReLU --------------------------------------------------------
__global__ __launch_bounds__(256) void finalize_kernel(const float* __restrict__ hout,
                                                       float* __restrict__ out) {
    int t = blockIdx.x * 256 + threadIdx.x;              // < N*64
    int i = t >> 6, o = t & 63;
    const float* p = hout + (size_t)i * 256 + o;
    float v = 0.25f * (p[0] + p[64] + p[128] + p[192]);
    out[t] = v > 0.f ? v : 0.f;
}

extern "C" void kernel_launch(void* const* d_in, const int* in_sizes, int n_in,
                              void* d_out, int out_size, void* d_ws, size_t ws_size,
                              hipStream_t stream) {
    const float* nfeats = (const float*)d_in[0];
    const float* reward = (const float*)d_in[1];
    const int*   src    = (const int*)  d_in[2];
    const int*   dst    = (const int*)  d_in[3];
    const float* W_ni   = (const float*)d_in[4];
    const float* W_nj   = (const float*)d_in[5];
    const float* W_fij  = (const float*)d_in[6];
    const float* W_node = (const float*)d_in[7];
    const float* bias   = (const float*)d_in[8];
    const float* attn   = (const float*)d_in[9];
    float* out = (float*)d_out;
    (void)in_sizes; (void)n_in; (void)out_size; (void)ws_size;

    char* ws = (char*)d_ws;
    size_t off = 0;
    float* Fbig  = (float*)(ws + off); off += (size_t)NN * HO * 4;   // 153.6 MB
    float* elog  = (float*)(ws + off); off += (size_t)EE * 4  * 4;   //  12.8 MB
    float* emax  = (float*)(ws + off); off += (size_t)NN * 4  * 4;
    float* denom = (float*)(ws + off); off += (size_t)NN * 4  * 4;
    float* hout  = (float*)(ws + off); off += (size_t)NN * 256 * 4;  //  51.2 MB
    unsigned short* wpack = (unsigned short*)(ws + off); off += (size_t)196608 * 2;
    float* wsum  = (float*)(ws + off); off += 256 * 4;

    pack_w_kernel<<<768, 256, 0, stream>>>(W_ni, W_nj, W_node, wpack);
    wsum_kernel<<<1, 256, 0, stream>>>(W_fij, wsum);
    init_kernel<<<(NN * 256) / 256, 256, 0, stream>>>(emax, denom, hout);
    gemm_kernel<<<dim3(NN / 16, 3), 256, 0, stream>>>(nfeats, wpack, Fbig);
    logits_kernel<<<EE / 8, 256, 0, stream>>>(Fbig, reward, src, dst, wsum, bias,
                                              attn, elog, emax);
    expsum_kernel<<<(EE * 4) / 256, 256, 0, stream>>>(elog, emax, dst, denom);
    aggregate_kernel<<<EE / 8, 256, 0, stream>>>(Fbig, elog, denom, src, dst, hout);
    finalize_kernel<<<(NN * 64) / 256, 256, 0, stream>>>(hout, out);
}
